// VolumnRenderer_33913061769778
// MI455X (gfx1250) — matrix operand order repaired and checked by reading, not compiled, not run on hardware
//
#include <hip/hip_runtime.h>
#include <math.h>

// ---------------- problem constants (match reference) ----------------
#define RESG     128
#define HIDDEN   128
#define PE_POS   10
#define PE_DIR   4
#define S_MAXS   192
#define STEP_SZ  0.027f
#define ERT      1e-4f
#define AABB_MIN_F (-1.5f)
#define AABB_MAX_F ( 1.5f)

typedef __attribute__((ext_vector_type(16))) _Float16 v16h;
typedef __attribute__((ext_vector_type(8)))  _Float16 v8h;
typedef __attribute__((ext_vector_type(8)))  float    v8f;

// ---------------- LDS layout (bytes) ----------------
// Weights stored TRANSPOSED: Wt[n][k] so B-matrix columns are contiguous in K.
#define OFF_W0   0                        // f16 [128][64]   (k: 63 -> pad 64)
#define OFF_W1   (OFF_W0 + 128*64*2)      // f16 [128][128]
#define OFF_W2   (OFF_W1 + 128*128*2)     // f16 [128][160]  (k: 155 -> pad 160)
#define OFF_HB   (OFF_W2 + 128*160*2)     // f16 [16][128] head B: col0=w_sigma, col1..3=w_rgb
#define OFF_STAGE (OFF_HB + 16*128*2)
// per-wave staging: bufA 16x64 f16, bufB 16x128 f16, bufC 16x160 f16, head 16x4 f32
#define STAGE_F16_ELEMS (16*64 + 16*128 + 16*160)            // 5632 f16 = 11264 B
#define STAGE_BYTES_PER_WAVE (STAGE_F16_ELEMS*2 + 16*4*4)    // + 256 B head = 11520 B
#define NWAVES 8
#define SMEM_BYTES (OFF_STAGE + NWAVES * STAGE_BYTES_PER_WAVE)   // ~182 KB (< 320 KB WGP LDS)

// ---------------- WMMA fragment helpers (CDNA5 16x16x32 f16 layouts) ----------------

// A (16x32 f16): lane L (half h = L>>4, m = L&15) holds K = {kc*32 + h*8 .. +7} and {+16..}
__device__ __forceinline__ v16h load_a_frag(const _Float16* tile, int ld, int kc, int lane) {
  const int m = lane & 15, h = lane >> 4;
  const _Float16* p = tile + m * ld + kc * 32 + h * 8;
  v8h lo = *(const v8h*)(p);
  v8h hi = *(const v8h*)(p + 16);
  v16h a;
#pragma unroll
  for (int i = 0; i < 8; ++i) { a[i] = lo[i]; a[i + 8] = hi[i]; }
  return a;
}

// B (32x16 f16): lane L holds column n = nt*16+(L&15), K = {kc*32 + (L>>4)*8 .. +7} and {+16..}
__device__ __forceinline__ v16h load_b_frag(const _Float16* wt, int ldk, int kc, int nt, int lane) {
  const int n = nt * 16 + (lane & 15), h = lane >> 4;
  const _Float16* p = wt + n * ldk + kc * 32 + h * 8;
  v8h lo = *(const v8h*)(p);
  v8h hi = *(const v8h*)(p + 16);
  v16h b;
#pragma unroll
  for (int i = 0; i < 8; ++i) { b[i] = lo[i]; b[i + 8] = hi[i]; }
  return b;
}

// D (16x16 f32, 8 VGPRs): VGPR v <-> row v + 8*(L>>4), col = L&15 within N-tile.
__device__ __forceinline__ void store_d_relu_f16(v8f d, _Float16* tile, int ld, int nt, int lane) {
  const int col = nt * 16 + (lane & 15), rbase = (lane >> 4) * 8;
#pragma unroll
  for (int v = 0; v < 8; ++v)
    tile[(rbase + v) * ld + col] = (_Float16)fmaxf(d[v], 0.0f);
}

// Head D store: keep only cols 0..3 (sigma / rgb) into [16][4] f32 buffer.
__device__ __forceinline__ void store_d_head(v8f d, float* hb, int lane) {
  const int col = lane & 15, rbase = (lane >> 4) * 8;
  if (col < 4) {
#pragma unroll
    for (int v = 0; v < 8; ++v) hb[(rbase + v) * 4 + col] = d[v];
  }
}

__device__ __forceinline__ v8f wmma_f16(v16h a, v16h b, v8f c) {
  return __builtin_amdgcn_wmma_f32_16x16x32_f16(false, a, false, b, (short)0, c, false, false);
}

// wave-local LDS RAW fence (CDNA5 split wait counters)
__device__ __forceinline__ void lds_wait() { asm volatile("s_wait_dscnt 0" ::: "memory"); }

__device__ __forceinline__ float sigmoidf_(float x) { return 1.0f / (1.0f + __expf(-x)); }

// ---------------- fused NeRF ray-march kernel ----------------
__global__ __launch_bounds__(256, 1)
void nerf_fused_march(const float* __restrict__ rays_o,
                      const float* __restrict__ viewdirs,
                      const unsigned char* __restrict__ occ,   // jax bool -> 1 byte
                      const float* __restrict__ gW0,   // [63][128]
                      const float* __restrict__ gW1,   // [128][128]
                      const float* __restrict__ gws,   // [128][1]
                      const float* __restrict__ gW2,   // [155][128]
                      const float* __restrict__ gwr,   // [128][3]
                      float* __restrict__ out_rgb,     // [R][3]
                      float* __restrict__ out_depth,   // [R]
                      float* __restrict__ out_opa,     // [R]
                      int R)
{
  extern __shared__ __align__(16) char smem[];
  _Float16* lW0 = (_Float16*)(smem + OFF_W0);
  _Float16* lW1 = (_Float16*)(smem + OFF_W1);
  _Float16* lW2 = (_Float16*)(smem + OFF_W2);
  _Float16* lHB = (_Float16*)(smem + OFF_HB);

  // ---- cooperative weight stage: global f32 -> LDS f16, transposed [n][k] ----
  for (int idx = threadIdx.x; idx < 128 * 64; idx += blockDim.x) {
    int n = idx >> 6, k = idx & 63;
    lW0[idx] = (_Float16)((k < 63) ? gW0[k * HIDDEN + n] : 0.0f);
  }
  for (int idx = threadIdx.x; idx < 128 * 128; idx += blockDim.x) {
    int n = idx >> 7, k = idx & 127;
    lW1[idx] = (_Float16)gW1[k * HIDDEN + n];
  }
  for (int idx = threadIdx.x; idx < 128 * 160; idx += blockDim.x) {
    int n = idx / 160, k = idx - n * 160;
    lW2[idx] = (_Float16)((k < 155) ? gW2[k * HIDDEN + n] : 0.0f);
  }
  // head B-tile: col 0 = w_sigma, cols 1..3 = w_rgb channels, cols 4..15 = 0
  for (int idx = threadIdx.x; idx < 16 * 128; idx += blockDim.x) {
    int n = idx >> 7, k = idx & 127;
    float v = 0.0f;
    if (n == 0)      v = gws[k];
    else if (n < 4)  v = gwr[k * 3 + (n - 1)];
    lHB[idx] = (_Float16)v;
  }
  __syncthreads();

  const int lane = threadIdx.x & 31;
  const int wave = threadIdx.x >> 5;
  const int wavesPerBlk = blockDim.x >> 5;
  const int totalWaves = gridDim.x * wavesPerBlk;

  char* stgb = smem + OFF_STAGE + wave * STAGE_BYTES_PER_WAVE;
  _Float16* bufA = (_Float16*)stgb;                 // 16x64  : pe_pos tile
  _Float16* bufB = bufA + 16 * 64;                  // 16x128 : h1 / h3 tile (reused)
  _Float16* bufC = bufB + 16 * 128;                 // 16x160 : [h2 | pe_dir] tile
  float*    hbuf = (float*)(stgb + STAGE_F16_ELEMS * 2);   // 16x4 f32 head outputs

  const float occ_scale = (float)RESG / (AABB_MAX_F - AABB_MIN_F);

  for (int ray = blockIdx.x * wavesPerBlk + wave; ray < R; ray += totalWaves) {
    // ---- ray setup (uniform across wave) ----
    float ox = rays_o[ray * 3 + 0], oy = rays_o[ray * 3 + 1], oz = rays_o[ray * 3 + 2];
    float dx = viewdirs[ray * 3 + 0], dy = viewdirs[ray * 3 + 1], dz = viewdirs[ray * 3 + 2];

    float cdx = (fabsf(dx) < 1e-9f) ? 1e-9f : dx;
    float cdy = (fabsf(dy) < 1e-9f) ? 1e-9f : dy;
    float cdz = (fabsf(dz) < 1e-9f) ? 1e-9f : dz;
    float ivx = 1.0f / cdx, ivy = 1.0f / cdy, ivz = 1.0f / cdz;
    float t0x = (AABB_MIN_F - ox) * ivx, t1x = (AABB_MAX_F - ox) * ivx;
    float t0y = (AABB_MIN_F - oy) * ivy, t1y = (AABB_MAX_F - oy) * ivy;
    float t0z = (AABB_MIN_F - oz) * ivz, t1z = (AABB_MAX_F - oz) * ivz;
    float tnear = fmaxf(fmaxf(fminf(t0x, t1x), fminf(t0y, t1y)), fminf(t0z, t1z));
    tnear = fmaxf(tnear, 0.0f);
    float tfar = fminf(fminf(fmaxf(t0x, t1x), fmaxf(t0y, t1y)), fmaxf(t0z, t1z));

    // ---- view-dir encoding: constant per ray; replicate into bufC cols 128..159 ----
    if (lane < 16) {
      _Float16* row = bufC + lane * 160 + 128;
      row[0] = (_Float16)dx; row[1] = (_Float16)dy; row[2] = (_Float16)dz;
      float f = 1.0f;
#pragma unroll
      for (int l = 0; l < PE_DIR; ++l) {
        row[3 + 6 * l + 0] = (_Float16)__sinf(dx * f);
        row[3 + 6 * l + 1] = (_Float16)__sinf(dy * f);
        row[3 + 6 * l + 2] = (_Float16)__sinf(dz * f);
        row[3 + 6 * l + 3] = (_Float16)__cosf(dx * f);
        row[3 + 6 * l + 4] = (_Float16)__cosf(dy * f);
        row[3 + 6 * l + 5] = (_Float16)__cosf(dz * f);
        f *= 2.0f;
      }
#pragma unroll
      for (int j = 27; j < 32; ++j) row[j] = (_Float16)0.0f;
    }

    float accR = 0.0f, accG = 0.0f, accB = 0.0f, accD = 0.0f, accO = 0.0f;
    float prev_T = 1.0f;

    for (int s0 = 0; s0 < S_MAXS; s0 += 16) {
      // uniform early exits: transmittance dead, or chunk fully past far plane
      if (prev_T <= ERT) break;
      if (tnear + STEP_SZ * (float)s0 + 0.5f * STEP_SZ >= tfar) break;

      // ---- per-sample: position encode + occupancy mask (lanes 0..15 = rows) ----
      float t_mid = 0.0f;
      int maskv = 0;
      if (lane < 16) {
        float fi = (float)(s0 + lane);
        float t_start = tnear + STEP_SZ * fi;
        t_mid = t_start + 0.5f * STEP_SZ;
        int inside = (t_mid < tfar) && (tfar > tnear);
        float px = ox + dx * t_mid, py = oy + dy * t_mid, pz = oz + dz * t_mid;
        int gx = min(max((int)((px - AABB_MIN_F) * occ_scale), 0), RESG - 1);
        int gy = min(max((int)((py - AABB_MIN_F) * occ_scale), 0), RESG - 1);
        int gz = min(max((int)((pz - AABB_MIN_F) * occ_scale), 0), RESG - 1);
        maskv = inside && (occ[(gx * RESG + gy) * RESG + gz] != 0);

        _Float16* row = bufA + lane * 64;
        row[0] = (_Float16)px; row[1] = (_Float16)py; row[2] = (_Float16)pz;
        float f = 1.0f;
#pragma unroll
        for (int l = 0; l < PE_POS; ++l) {
          row[3 + 6 * l + 0] = (_Float16)__sinf(px * f);
          row[3 + 6 * l + 1] = (_Float16)__sinf(py * f);
          row[3 + 6 * l + 2] = (_Float16)__sinf(pz * f);
          row[3 + 6 * l + 3] = (_Float16)__cosf(px * f);
          row[3 + 6 * l + 4] = (_Float16)__cosf(py * f);
          row[3 + 6 * l + 5] = (_Float16)__cosf(pz * f);
          f *= 2.0f;
        }
        row[63] = (_Float16)0.0f;
      }
      lds_wait();

      // ---- layer 0: h1 = relu(pe_p @ W0)  [16x64]x[64x128]; A-frags hoisted ----
      {
        v16h a0, a1;
        a0 = load_a_frag(bufA, 64, 0, lane);
        a1 = load_a_frag(bufA, 64, 1, lane);
        for (int nt = 0; nt < 8; ++nt) {
          v8f acc = {};
          acc = wmma_f16(a0, load_b_frag(lW0, 64, 0, nt, lane), acc);
          acc = wmma_f16(a1, load_b_frag(lW0, 64, 1, nt, lane), acc);
          store_d_relu_f16(acc, bufB, 128, nt, lane);
        }
      }
      lds_wait();

      // ---- layer 1: h2 = relu(h1 @ W1)  -> bufC cols 0..127 ----
      {
        v16h a[4];
#pragma unroll
        for (int kc = 0; kc < 4; ++kc) a[kc] = load_a_frag(bufB, 128, kc, lane);
        for (int nt = 0; nt < 8; ++nt) {
          v8f acc = {};
#pragma unroll
          for (int kc = 0; kc < 4; ++kc)
            acc = wmma_f16(a[kc], load_b_frag(lW1, 128, kc, nt, lane), acc);
          store_d_relu_f16(acc, bufC, 160, nt, lane);
        }
      }
      lds_wait();

      // ---- A-frags of [h2 | pe_d] (16x160, 5 K-chunks); reused by sigma head + layer 2 ----
      v16h a2[5];
#pragma unroll
      for (int kc = 0; kc < 5; ++kc) a2[kc] = load_a_frag(bufC, 160, kc, lane);

      // sigma head: h2 @ [w_sigma|w_rgb] tile, keep col 0 (K = 128 -> 4 chunks)
      {
        v8f acc = {};
#pragma unroll
        for (int kc = 0; kc < 4; ++kc)
          acc = wmma_f16(a2[kc], load_b_frag(lHB, 128, kc, 0, lane), acc);
        store_d_head(acc, hbuf, lane);
      }

      // layer 2: h3 = relu([h2|pe_d] @ W2) -> bufB
      for (int nt = 0; nt < 8; ++nt) {
        v8f acc = {};
#pragma unroll
        for (int kc = 0; kc < 5; ++kc)
          acc = wmma_f16(a2[kc], load_b_frag(lW2, 160, kc, nt, lane), acc);
        store_d_relu_f16(acc, bufB, 128, nt, lane);
      }
      lds_wait();

      // read sigma (col 0) for this lane's row before head buffer is reused for rgb
      float sigma = 0.0f;
      if (lane < 16) sigma = hbuf[lane * 4 + 0];

      // rgb head: h3 @ [w_sigma|w_rgb] tile, keep cols 1..3 (K = 128 -> 4 chunks)
      {
        v16h a3[4];
#pragma unroll
        for (int kc = 0; kc < 4; ++kc) a3[kc] = load_a_frag(bufB, 128, kc, lane);
        v8f acc = {};
#pragma unroll
        for (int kc = 0; kc < 4; ++kc)
          acc = wmma_f16(a3[kc], load_b_frag(lHB, 128, kc, 0, lane), acc);
        store_d_head(acc, hbuf, lane);
      }
      lds_wait();

      float r0 = 0.0f, r1 = 0.0f, r2 = 0.0f;
      if (lane < 16) {
        r0 = sigmoidf_(hbuf[lane * 4 + 1]);
        r1 = sigmoidf_(hbuf[lane * 4 + 2]);
        r2 = sigmoidf_(hbuf[lane * 4 + 3]);
      }

      // ---- alpha compositing: 16-lane prefix product of (1-alpha+1e-10) ----
      float alpha = 0.0f;
      if (lane < 16)
        alpha = (1.0f - __expf(-fmaxf(sigma, 0.0f) * STEP_SZ)) * (maskv ? 1.0f : 0.0f);
      float fac = 1.0f - alpha + 1e-10f;
      float p = fac;
#pragma unroll
      for (int off = 1; off < 16; off <<= 1) {
        float q = __shfl_up(p, off);
        if ((lane & 15) >= off) p *= q;
      }
      float sh = __shfl_up(p, 1);
      float excl = ((lane & 15) == 0) ? 1.0f : sh;        // exclusive cumprod within chunk
      float T = prev_T * excl;
      T = (T > ERT) ? T : 0.0f;                            // early-ray-termination zeroing
      float w = T * alpha;

      float pr = 0.0f, pg = 0.0f, pb = 0.0f, pd = 0.0f, po = 0.0f;
      if (lane < 16) { pr = w * r0; pg = w * r1; pb = w * r2; pd = w * t_mid; po = w; }
#pragma unroll
      for (int off = 16; off > 0; off >>= 1) {
        pr += __shfl_xor(pr, off); pg += __shfl_xor(pg, off); pb += __shfl_xor(pb, off);
        pd += __shfl_xor(pd, off); po += __shfl_xor(po, off);
      }
      accR += pr; accG += pg; accB += pb; accD += pd; accO += po;

      prev_T *= __shfl(p, 15);                             // carry transmittance (uniform)
    }

    if (lane == 0) {
      float bg = 1.0f - accO;                              // white background
      out_rgb[ray * 3 + 0] = accR + bg;
      out_rgb[ray * 3 + 1] = accG + bg;
      out_rgb[ray * 3 + 2] = accB + bg;
      out_depth[ray] = accD;
      out_opa[ray]   = accO;
    }
  }
}

extern "C" void kernel_launch(void* const* d_in, const int* in_sizes, int n_in,
                              void* d_out, int out_size, void* d_ws, size_t ws_size,
                              hipStream_t stream) {
  const float*         rays_o   = (const float*)d_in[0];
  const float*         viewdirs = (const float*)d_in[1];
  const unsigned char* occ      = (const unsigned char*)d_in[2];  // jax bool grid
  const float*         W0       = (const float*)d_in[3];
  const float*         W1       = (const float*)d_in[4];
  const float*         w_sigma  = (const float*)d_in[5];
  const float*         W2       = (const float*)d_in[6];
  const float*         w_rgb    = (const float*)d_in[7];

  const int R = in_sizes[0] / 3;                  // 2048 rays
  float* out_rgb   = (float*)d_out;               // [R,3]
  float* out_depth = out_rgb + (size_t)R * 3;     // [R,1]
  float* out_opa   = out_depth + (size_t)R;       // [R,1]

  const int block = 32 * NWAVES;                  // 8 wave32 per block
  const int grid  = (R + NWAVES - 1) / NWAVES;    // 1 ray per wave
  hipLaunchKernelGGL(nerf_fused_march, dim3(grid), dim3(block), SMEM_BYTES, stream,
                     rays_o, viewdirs, occ, W0, W1, w_sigma, W2, w_rgb,
                     out_rgb, out_depth, out_opa, R);
}
